// Tree_Net_9826885173865
// MI455X (gfx1250) — matrix operand order repaired and checked by reading, not compiled, not run on hardware
//
#include <hip/hip_runtime.h>

// Problem constants (from the reference)
#define B_    128
#define L_    512
#define N_    1023
#define T_    511
#define D_    256
#define V_    50000
#define C_    511
#define CPAD_ 512
#define EPS_  1e-6f

typedef __attribute__((ext_vector_type(16))) _Float16 v16h;
typedef __attribute__((ext_vector_type(8)))  _Float16 v8h;
typedef __attribute__((ext_vector_type(8)))  float    v8f;

// ---------------------------------------------------------------------------
// 1) Scatter normalized embedding rows into vec (B,N,D). vec pre-zeroed.
//    One 256-thread block per (b,l) pair; slot indices are unique per batch.
// ---------------------------------------------------------------------------
__global__ void scatter_leaves(float* __restrict__ vec,
                               const int* __restrict__ leaf_ids,   // (B,L,2)
                               const int* __restrict__ mask,       // (B,L)
                               const float* __restrict__ emb) {    // (V,D)
    int bl = blockIdx.x;
    int b  = bl / L_;
    int l  = bl % L_;
    if (mask[b * L_ + l] == 0) return;
    int node  = leaf_ids[(b * L_ + l) * 2 + 0];
    int vocab = leaf_ids[(b * L_ + l) * 2 + 1];
    int d = threadIdx.x;
    float x = emb[(size_t)vocab * D_ + d];
    __shared__ float red[D_];
    red[d] = x * x;
    __syncthreads();
    for (int s = D_ / 2; s > 0; s >>= 1) {
        if (d < s) red[d] += red[d + s];
        __syncthreads();
    }
    float inv = 1.0f / (sqrtf(red[0]) + EPS_);
    vec[((size_t)b * N_ + node) * D_ + d] = x * inv;
}

// ---------------------------------------------------------------------------
// 2) Sequential compose scan: one block per batch, 256 threads = D lanes.
//    Circular correlation done directly in LDS (D^2 = 64K MACs / step).
// ---------------------------------------------------------------------------
__global__ void compose_kernel(float* __restrict__ vec,
                               const int* __restrict__ comp) {      // (B,T,4)
    int b = blockIdx.x;
    int d = threadIdx.x;                 // 0..255
    float* vb = vec + (size_t)b * N_ * D_;
    const int* cb = comp + (size_t)b * T_ * 4;
    __shared__ float lv[D_], rv[D_], red[D_];
    for (int t = 0; t < T_; ++t) {
        int ty = cb[t * 4 + 0];
        int p  = cb[t * 4 + 1];
        int l  = cb[t * 4 + 2];
        int r  = cb[t * 4 + 3];
        if (ty == 1) {
            // copy left child to parent (per-lane element, no cross-lane hazard)
            float x = vb[(size_t)l * D_ + d];
            vb[(size_t)p * D_ + d] = x;
        } else if (ty == 2) {
            lv[d] = vb[(size_t)l * D_ + d];
            rv[d] = vb[(size_t)r * D_ + d];
            __syncthreads();
            float acc = 0.0f;
#pragma unroll 8
            for (int j = 0; j < D_; ++j)
                acc = fmaf(lv[j], rv[(j + d) & (D_ - 1)], acc);
            red[d] = acc * acc;
            __syncthreads();
            for (int s = D_ / 2; s > 0; s >>= 1) {
                if (d < s) red[d] += red[d + s];
                __syncthreads();
            }
            float inv = 1.0f / (sqrtf(red[0]) + EPS_);
            vb[(size_t)p * D_ + d] = acc * inv;
        }
        // ty==0: writing vb[b,0]=vb[b,0] is a no-op in the reference
        __syncthreads();   // publish this step's global writes to the block
    }
}

// ---------------------------------------------------------------------------
// 3) Convert vec f32 -> f16 (A operand of the GEMM)
// ---------------------------------------------------------------------------
__global__ void convert_f32_to_f16(const float* __restrict__ src,
                                   _Float16* __restrict__ dst, size_t n) {
    size_t i = (size_t)blockIdx.x * blockDim.x + threadIdx.x;
    if (i < n) dst[i] = (_Float16)src[i];
}

// 3b) Weight: convert + zero-pad C 511 -> 512 so B-fragment loads are unguarded
__global__ void prep_weight(const float* __restrict__ w,  // (C,D)
                            _Float16* __restrict__ wh) {  // (CPAD,D)
    int c = blockIdx.x, d = threadIdx.x;
    wh[c * D_ + d] = (c < C_) ? (_Float16)w[c * D_ + d] : (_Float16)0.0f;
}

// ---------------------------------------------------------------------------
// 4) WMMA GEMM: out(BN x 511) = vecH(BN x 256) * WH^T(256 x 512) + bias
//    One wave per 16-row M-tile; A K-fragments kept in registers (read once),
//    loop over all 32 N-tiles of the L2-resident weight.
//    A/B use contiguous K order (K = 16*g + s, same for both); f32 C/D layout per CDNA5 ISA 7.12.2.
// ---------------------------------------------------------------------------
__global__ void __launch_bounds__(256)
gemm_wmma(const _Float16* __restrict__ vecH,   // (B*N, 256)
          const _Float16* __restrict__ WH,     // (512, 256)
          const float* __restrict__ bias,      // (511)
          float* __restrict__ out) {           // (B*N, 511)
    const int lane  = threadIdx.x & 31;
    const int wave  = threadIdx.x >> 5;            // 0..7
    const int mtile = blockIdx.x * 8 + wave;       // 0..8183
    const int m0    = mtile * 16;
    const int g     = lane >> 4;                   // lane half
    const int mr    = lane & 15;                   // row (A) / col (B,D)

    // --- A fragments: lane holds row M=mr; lane-half g selects K 0-15
    //     (g=0) or K 16-31 (g=1) within each 32-wide K block.
    v16h aF[8];
    const _Float16* arow = vecH + (size_t)(m0 + mr) * D_;
#pragma unroll
    for (int kk = 0; kk < 8; ++kk) {
        const int k0 = kk * 32;
        v16h a = *(const v16h*)(arow + k0 + g * 16);      // K = k0 + g*16 .. +15
        aF[kk] = a;
    }

    // --- Loop over 32 N-tiles of the padded weight
    for (int ct = 0; ct < 32; ++ct) {
        const int c0 = ct * 16;
        v8f acc = {};
        // B fragment: lane holds col N=mr; lane half g selects K 0-15 / 16-31
        const _Float16* wrow = WH + (size_t)(c0 + mr) * D_ + g * 16;
#pragma unroll
        for (int kk = 0; kk < 8; ++kk) {
            v16h bf = *(const v16h*)(wrow + kk * 32);
            acc = __builtin_amdgcn_wmma_f32_16x16x32_f16(
                /*neg_a=*/false, aF[kk], /*neg_b=*/false, bf,
                /*c_mod=*/(short)0, acc, /*reuse_a=*/false, /*reuse_b=*/false);
        }
        // D layout: lane col N=mr, VGPR v holds row M = v + 8*g
        const int col = c0 + mr;
        if (col < C_) {
            const float bv = bias[col];
#pragma unroll
            for (int v = 0; v < 8; ++v) {
                const int row = m0 + v + 8 * g;
                out[(size_t)row * C_ + col] = acc[v] + bv;
            }
        }
    }
}

// ---------------------------------------------------------------------------
extern "C" void kernel_launch(void* const* d_in, const int* in_sizes, int n_in,
                              void* d_out, int out_size, void* d_ws, size_t ws_size,
                              hipStream_t stream) {
    // inputs: 0 num_node (unused), 1 leaf_content_id, 2 content_mask,
    //         3 composition_info, 4 emb_weight, 5 lin_weight, 6 lin_bias
    const int*   leaf_ids = (const int*)d_in[1];
    const int*   mask     = (const int*)d_in[2];
    const int*   comp     = (const int*)d_in[3];
    const float* emb      = (const float*)d_in[4];
    const float* lin_w    = (const float*)d_in[5];
    const float* lin_b    = (const float*)d_in[6];
    float* out = (float*)d_out;

    // workspace layout (all sections 256B aligned by construction)
    const size_t vecElems  = (size_t)B_ * N_ * D_;              // 33,521,664
    const size_t vecBytes  = vecElems * sizeof(float);          // ~134 MB
    const size_t vecHBytes = vecElems * sizeof(_Float16);       // ~67 MB
    char* ws = (char*)d_ws;
    float*    vec  = (float*)ws;
    _Float16* vecH = (_Float16*)(ws + vecBytes);
    _Float16* wh   = (_Float16*)(ws + vecBytes + vecHBytes);

    // 1) zero vec, scatter normalized leaf embeddings
    hipMemsetAsync(vec, 0, vecBytes, stream);
    scatter_leaves<<<B_ * L_, D_, 0, stream>>>(vec, leaf_ids, mask, emb);

    // 2) sequential tree composition (one block per batch)
    compose_kernel<<<B_, D_, 0, stream>>>(vec, comp);

    // 3) precision convert for WMMA operands
    convert_f32_to_f16<<<(int)(vecElems / 256), 256, 0, stream>>>(vec, vecH, vecElems);
    prep_weight<<<CPAD_, D_, 0, stream>>>(lin_w, wh);

    // 4) WMMA projection: 8184 M-tiles, 8 waves/block
    gemm_wmma<<<(B_ * N_ / 16) / 8, 256, 0, stream>>>(vecH, wh, lin_b, out);
}
